// MOE_55834574848057
// MI455X (gfx1250) — matrix operand order repaired and checked by reading, not compiled, not run on hardware
//
#include <hip/hip_runtime.h>
#include <hip/hip_bf16.h>

typedef __attribute__((ext_vector_type(16))) _Float16 v16h;
typedef __attribute__((ext_vector_type(8)))  _Float16 v8h;
typedef __attribute__((ext_vector_type(8)))  float    v8f;

#define B_SZ 16384
#define IN_SZ 256
#define E_SZ 32
#define OUT_SZ 128

// per-expert repacked weight block: 81920 halves
// layer offsets (halves): W1 0, W2 16384, W3 32768, W4 65536, W5 73728
#define EXP_W 81920

// LDS byte layout (dynamic shared only => offsets are absolute LDS addresses)
#define WLDS_BYTE 0
#define XT_BYTE   163840            // 2 x (4 x 16x264 f16) = 2 x 33792
#define BUFA_BYTE (163840 + 2 * 33792)          // 231424, 4 x 16x136 (17408 B)
#define BUFB_BYTE (231424 + 17408)              // 248832, 4 x 16x264 (33792 B)
#define SMEM_BYTES (248832 + 33792)             // 282624

// ---------------------------------------------------------------------------
__device__ __forceinline__ void async_load_b128(unsigned int ldsOff, unsigned long long gaddr) {
    asm volatile("global_load_async_to_lds_b128 %0, %1, off"
                 :: "v"(ldsOff), "v"(gaddr) : "memory");
}
__device__ __forceinline__ void wait_asynccnt0() {
    asm volatile("s_wait_asynccnt 0" ::: "memory");
}

// ---------------------------------------------------------------------------
__global__ void zero_out_kernel(float4* __restrict__ out) {
    out[(size_t)blockIdx.x * 256 + threadIdx.x] = make_float4(0.f, 0.f, 0.f, 0.f);
}

// x fp32 -> f16
__global__ void cvt_x_kernel(const float* __restrict__ src, _Float16* __restrict__ dst, int total) {
    int idx = blockIdx.x * 256 + threadIdx.x;
    if (idx < total) dst[idx] = (_Float16)src[idx];
}

// ---------------------------------------------------------------------------
// Repack W[e][K][N] fp32 -> f16 B-fragment tiles (512 halves per 32x16 tile).
// Within a tile: lane's K-lo 8 halves at [lane*8], K-hi 8 halves at [256+lane*8].
__global__ void repack_w_kernel(const float* __restrict__ src, _Float16* __restrict__ dst,
                                int K, int N, int layerOff, int total) {
    int idx = blockIdx.x * 256 + threadIdx.x;
    if (idx >= total) return;
    int n = idx % N;
    int k = (idx / N) % K;
    int e = idx / (N * K);
    int nt = n >> 4, nl = n & 15;
    int kt = k >> 5, kk = k & 31;
    int lane = (((kk >> 3) & 1) << 4) | nl;
    int j = (kk & 7) | ((kk >> 4) << 3);
    int KT = K >> 5;
    int pos = (j < 8) ? (lane * 8 + j) : (256 + lane * 8 + (j - 8));
    size_t d = (size_t)e * EXP_W + layerOff + ((size_t)(nt * KT + kt)) * 512 + pos;
    dst[d] = (_Float16)src[idx];
}

// ---------------------------------------------------------------------------
// Gate: g = softmax(x @ Wg + bg) over E=32 (one wave per row, lane = expert)
__global__ __launch_bounds__(256) void gate_kernel(const float* __restrict__ x,
                                                   const float* __restrict__ Wg,
                                                   const float* __restrict__ bg,
                                                   float* __restrict__ g) {
    int lane = threadIdx.x & 31;
    int wave = threadIdx.x >> 5;
    int row = blockIdx.x * 8 + wave;
    const float* xr = x + (size_t)row * IN_SZ;
    float acc = bg[lane];
    for (int i = 0; i < IN_SZ; ++i) acc += xr[i] * Wg[i * E_SZ + lane];
    float m = acc;
    for (int off = 16; off > 0; off >>= 1) m = fmaxf(m, __shfl_xor(m, off, 32));
    float p = __expf(acc - m);
    float s = p;
    for (int off = 16; off > 0; off >>= 1) s += __shfl_xor(s, off, 32);
    g[(size_t)row * E_SZ + lane] = p / s;
}

// ---------------------------------------------------------------------------
__device__ __forceinline__ v16h load_frag16(const _Float16* p, int hiOff) {
    v8h lo = *(const v8h*)(p);
    v8h hi = *(const v8h*)(p + hiOff);
    return __builtin_shufflevector(lo, hi, 0,1,2,3,4,5,6,7,8,9,10,11,12,13,14,15);
}

// Async prefetch of 64 rows of x (f16) into xt buffer `parity`
__device__ __forceinline__ void prefetch_x(const _Float16* __restrict__ xsrc, int parity, int tid) {
#pragma unroll
    for (int kk = 0; kk < 8; ++kk) {
        int i = tid + kk * 256;
        int r = i >> 5, seg = i & 31;
        unsigned int lo = XT_BYTE + parity * 33792 +
                          (((r >> 4) * (16 * 264) + (r & 15) * 264 + seg * 8) << 1);
        unsigned long long ga = (unsigned long long)(uintptr_t)(xsrc + r * IN_SZ + seg * 8);
        async_load_b128(lo, ga);
    }
}

// Cooperative layer over a 64-row group (4 row-tiles in LDS, stride INST halves).
template <int KT, int NT, int RW, int NU, int INST, int OUTST>
__device__ __forceinline__ void coop_layer(const _Float16* inBase,
                                           const _Float16* wl,
                                           const float* __restrict__ bias,
                                           _Float16* outBase,
                                           int lane, int rt0, int nt0) {
    const int row = lane & 15;
    const int kh = (lane >> 4) << 3;
    const int lanew = lane << 3;
    const int rhalf = (lane >> 4) << 3;
    v8f c[RW][NU];
#pragma unroll
    for (int r = 0; r < RW; ++r)
#pragma unroll
        for (int u = 0; u < NU; ++u) c[r][u] = (v8f){};
#pragma unroll
    for (int kt = 0; kt < KT; ++kt) {
        v16h a[RW];
#pragma unroll
        for (int r = 0; r < RW; ++r)
            a[r] = load_frag16(inBase + (rt0 + r) * 16 * INST + row * INST + kt * 32 + kh, 16);
        v16h b[NU];
#pragma unroll
        for (int u = 0; u < NU; ++u)
            b[u] = load_frag16(wl + ((nt0 + u) * KT + kt) * 512 + lanew, 256);
#pragma unroll
        for (int r = 0; r < RW; ++r)
#pragma unroll
            for (int u = 0; u < NU; ++u)
                c[r][u] = __builtin_amdgcn_wmma_f32_16x16x32_f16(false, a[r], false, b[u],
                                                                 (short)0, c[r][u], false, false);
    }
#pragma unroll
    for (int r = 0; r < RW; ++r)
#pragma unroll
        for (int u = 0; u < NU; ++u) {
            float bv = bias[(nt0 + u) * 16 + row];
#pragma unroll
            for (int v = 0; v < 8; ++v) {
                float t = fmaxf(c[r][u][v] + bv, 0.0f);
                outBase[(rt0 + r) * 16 * OUTST + (rhalf + v) * OUTST + (nt0 + u) * 16 + row] =
                    (_Float16)t;
            }
        }
}

// ---------------------------------------------------------------------------
// Expert-stationary main kernel: block = 8 waves, expert weights resident in LDS,
// 512 rows per block in 64-row cooperative groups; async double-buffered x staging.
__global__ __launch_bounds__(256) void moe_main_kernel(
    const _Float16* __restrict__ xh, const float* __restrict__ g,
    const _Float16* __restrict__ wall,
    const float* __restrict__ b1, const float* __restrict__ b2,
    const float* __restrict__ b3, const float* __restrict__ b4,
    const float* __restrict__ b5, float* __restrict__ out) {
    extern __shared__ __align__(16) char smem[];
    _Float16* wlds  = (_Float16*)(smem + WLDS_BYTE);
    _Float16* xtab  = (_Float16*)(smem + XT_BYTE);      // two 16896-half buffers
    _Float16* bufA  = (_Float16*)(smem + BUFA_BYTE);
    _Float16* bufB  = (_Float16*)(smem + BUFB_BYTE);

    const int tid = threadIdx.x;
    const int lane = tid & 31;
    const int w = tid >> 5;
    const int e = blockIdx.x & 31;
    const int blockRowBase = (blockIdx.x >> 5) * 512;

    // Async-stage: x row-group 0 and this expert's full weight set (no VGPR round-trip)
    prefetch_x(xh + (size_t)blockRowBase * IN_SZ, 0, tid);
    {
        const _Float16* wsrc = wall + (size_t)e * EXP_W;
#pragma unroll 1
        for (int kk = 0; kk < 40; ++kk) {
            int i = tid + kk * 256;
            async_load_b128((unsigned int)(WLDS_BYTE + i * 16),
                            (unsigned long long)(uintptr_t)(wsrc + i * 8));
        }
    }
    const float* b1e = b1 + e * 64;
    const float* b2e = b2 + e * 256;
    const float* b3e = b3 + e * 128;
    const float* b4e = b4 + e * 64;
    const float* b5e = b5 + e * 128;
    wait_asynccnt0();
    __syncthreads();

#pragma unroll 1
    for (int rg = 0; rg < 8; ++rg) {
        const int rgbase = blockRowBase + rg * 64;
        const _Float16* xt = xtab + (rg & 1) * 16896;
        // L1: x(64x256) @ W1 -> h1(64x64)
        coop_layer<8, 4, 1, 2, 264, 136>(xt, wlds, b1e, bufA, lane, w >> 1, 2 * (w & 1));
        // prefetch next row-group's x into the other buffer while we compute
        if (rg < 7) prefetch_x(xh + (size_t)(rgbase + 64) * IN_SZ, (rg + 1) & 1, tid);
        __syncthreads();
        // L2: h1 @ W2 -> h2(64x256)
        coop_layer<2, 16, 2, 4, 136, 264>(bufA, wlds + 16384, b2e, bufB, lane, 2 * (w & 1), 4 * (w >> 1));
        __syncthreads();
        // L3: h2 @ W3 -> h3(64x128)
        coop_layer<8, 8, 2, 2, 264, 136>(bufB, wlds + 32768, b3e, bufA, lane, 2 * (w & 1), 2 * (w >> 1));
        __syncthreads();
        // L4: h3 @ W4 -> h4(64x64)
        coop_layer<4, 4, 1, 2, 136, 264>(bufA, wlds + 65536, b4e, bufB, lane, w >> 1, 2 * (w & 1));
        __syncthreads();
        // L5: h4 @ W5 + b5, gate-weighted atomic accumulate into out
        {
            const int row = lane & 15;
            const int kh = (lane >> 4) << 3;
            const int lanew = lane << 3;
            const int rhalf = (lane >> 4) << 3;
            const int rt0 = 2 * (w & 1), nt0 = 2 * (w >> 1);
            const _Float16* w5l = wlds + 73728;
            v8f c[2][2];
#pragma unroll
            for (int r = 0; r < 2; ++r)
#pragma unroll
                for (int u = 0; u < 2; ++u) c[r][u] = (v8f){};
#pragma unroll
            for (int kt = 0; kt < 2; ++kt) {
                v16h a[2];
#pragma unroll
                for (int r = 0; r < 2; ++r)
                    a[r] = load_frag16(bufB + (rt0 + r) * (16 * 264) + row * 264 + kt * 32 + kh, 16);
                v16h b[2];
#pragma unroll
                for (int u = 0; u < 2; ++u)
                    b[u] = load_frag16(w5l + ((nt0 + u) * 2 + kt) * 512 + lanew, 256);
#pragma unroll
                for (int r = 0; r < 2; ++r)
#pragma unroll
                    for (int u = 0; u < 2; ++u)
                        c[r][u] = __builtin_amdgcn_wmma_f32_16x16x32_f16(false, a[r], false, b[u],
                                                                         (short)0, c[r][u], false, false);
            }
#pragma unroll
            for (int r = 0; r < 2; ++r) {
                const int rbase = rgbase + (rt0 + r) * 16 + rhalf;
                float gv[8];
#pragma unroll
                for (int v = 0; v < 8; ++v) gv[v] = g[(size_t)(rbase + v) * E_SZ + e];
#pragma unroll
                for (int u = 0; u < 2; ++u) {
                    float bv = b5e[(nt0 + u) * 16 + row];
#pragma unroll
                    for (int v = 0; v < 8; ++v) {
                        float val = gv[v] * (c[r][u][v] + bv);
                        float* p = out + (size_t)(rbase + v) * OUT_SZ + (nt0 + u) * 16 + row;
                        __hip_atomic_fetch_add(p, val, __ATOMIC_RELAXED, __HIP_MEMORY_SCOPE_AGENT);
                    }
                }
            }
        }
        wait_asynccnt0();
        __syncthreads();
    }
}

// ---------------------------------------------------------------------------
extern "C" void kernel_launch(void* const* d_in, const int* in_sizes, int n_in,
                              void* d_out, int out_size, void* d_ws, size_t ws_size,
                              hipStream_t stream) {
    const float* x  = (const float*)d_in[0];
    const float* Wg = (const float*)d_in[1];
    const float* bg = (const float*)d_in[2];
    const float* W1 = (const float*)d_in[3];
    const float* b1 = (const float*)d_in[4];
    const float* W2 = (const float*)d_in[5];
    const float* b2 = (const float*)d_in[6];
    const float* W3 = (const float*)d_in[7];
    const float* b3 = (const float*)d_in[8];
    const float* W4 = (const float*)d_in[9];
    const float* b4 = (const float*)d_in[10];
    const float* W5 = (const float*)d_in[11];
    const float* b5 = (const float*)d_in[12];
    float* out = (float*)d_out;

    char* ws = (char*)d_ws;
    _Float16* xh   = (_Float16*)(ws);               // 8 MB
    float*    g    = (float*)(ws + 8388608);        // 2 MB
    _Float16* wall = (_Float16*)(ws + 10485760);    // 32 * 81920 halves = 5 MB

    hipFuncSetAttribute((const void*)moe_main_kernel,
                        hipFuncAttributeMaxDynamicSharedMemorySize, SMEM_BYTES);

    zero_out_kernel<<<(B_SZ * OUT_SZ / 4) / 256, 256, 0, stream>>>((float4*)out);
    cvt_x_kernel<<<(B_SZ * IN_SZ) / 256, 256, 0, stream>>>(x, xh, B_SZ * IN_SZ);
    repack_w_kernel<<<(E_SZ * 256 * 64) / 256, 256, 0, stream>>>(W1, wall, 256, 64, 0, E_SZ * 256 * 64);
    repack_w_kernel<<<(E_SZ * 64 * 256) / 256, 256, 0, stream>>>(W2, wall, 64, 256, 16384, E_SZ * 64 * 256);
    repack_w_kernel<<<(E_SZ * 256 * 128) / 256, 256, 0, stream>>>(W3, wall, 256, 128, 32768, E_SZ * 256 * 128);
    repack_w_kernel<<<(E_SZ * 128 * 64) / 256, 256, 0, stream>>>(W4, wall, 128, 64, 65536, E_SZ * 128 * 64);
    repack_w_kernel<<<(E_SZ * 64 * 128) / 256, 256, 0, stream>>>(W5, wall, 64, 128, 73728, E_SZ * 64 * 128);
    gate_kernel<<<B_SZ / 8, 256, 0, stream>>>(x, Wg, bg, g);
    moe_main_kernel<<<E_SZ * (B_SZ / 512), 256, SMEM_BYTES, stream>>>(
        xh, g, wall, b1, b2, b3, b4, b5, out);
}
